// TopkPointExtractor_29540785062397
// MI455X (gfx1250) — compile-verified, matrix operand order checked by reading, not run on hardware
//
#include <hip/hip_runtime.h>
#include <hip/hip_bf16.h>

// ---------------- problem constants (from reference) ----------------
constexpr int BATCH = 16;
constexpr int CIN   = 1024;   // NBFEAT
constexpr int ENC   = 64;
constexpr int KTOP  = 1024;
constexpr int HFULL = 62, WFULL = 62;
constexpr int CROP  = 3;
constexpr int HC = 56, WC = 56;
constexpr int NPIX  = HC * WC;        // 3136
constexpr int NTILE = NPIX / 16;      // 196 position-tiles per batch
constexpr int NSORT = 4096;           // next pow2 >= NPIX

// LDS pitch for the resident bf16 weight slab: 1040 elements = 2080 B.
// 2080 mod 256 = 32 -> consecutive M rows shift by 8 banks' worth of bytes;
// the 16 lanes' 16B fragments spread across all 64 banks (conflict-free),
// and every fragment stays 16B-aligned for ds_load_b128.
constexpr int APITCH = 1040;

typedef __attribute__((ext_vector_type(16))) __bf16 v16bf;
typedef __attribute__((ext_vector_type(8)))  float  v8f;

// =====================================================================
// K1: fused WMMA GEMM: pfm = w1 @ fm (+b1), depth head, x = sum(pfm^2)
//   grid: 392 blocks x 256 threads (8 waves); each wave owns one
//   16-position tile; 3136 tiles total = 392*8.
//   Full bf16 weight matrix (65 x 1024, ~135 KB) resident in LDS:
//   staged once, then the K-loop runs with NO block barriers.
// A/B fragment layout per CDNA5 ISA 7.12.2 (16-bit, 16x32):
//   lane L: M (or N) = L&15 ; element j (0..15) maps to
//   K(j,L) = (j<8 ? 0 : 16) + 8*(L>>4) + (j&7)
// C/D layout: VGPR r, lane L -> (M = r + 8*(L>>4), N = L&15)
// =====================================================================
__global__ __launch_bounds__(256)
void k_gemm_fused(const float* __restrict__ img,   // (B, CIN, 62, 62)
                  const float* __restrict__ w1,    // (ENC, CIN)
                  const float* __restrict__ b1,    // (ENC)
                  const float* __restrict__ wdep,  // (1, CIN)
                  const float* __restrict__ bdep,  // (1)
                  float* __restrict__ pfm,         // (B, ENC, NPIX)
                  float* __restrict__ depthMap,    // (B, NPIX)
                  float* __restrict__ xmap)        // (B, NPIX)
{
    __shared__ __bf16 a_lds[65 * APITCH];   // rows 0..63 = w1, row 64 = w_depth

    const int tid   = threadIdx.x;
    const int lane  = tid & 31;
    const int wave  = tid >> 5;                       // 0..7
    const int gtile = blockIdx.x * 8 + wave;          // 0..3135
    const int b     = gtile / NTILE;
    const int tile  = gtile % NTILE;
    const int pos   = tile * 16 + (lane & 15);        // this lane's column N
    const int py    = pos / WC, px = pos % WC;
    const long imgBase = ((long)b * CIN) * (WFULL * WFULL)
                       + (long)(py + CROP) * WFULL + (px + CROP);
    const int khalf = (lane >> 4) * 8;                // 0 or 8
    const int m0    = lane & 15;

    // ---- one-time cooperative A staging: (64+1) x 1024, f32 -> bf16 ----
    for (int i = tid; i < 65 * CIN; i += 256) {
        const int row = i >> 10, col = i & 1023;      // coalesced along col
        const float v = (row < ENC) ? w1[row * CIN + col] : wdep[col];
        a_lds[row * APITCH + col] = (__bf16)v;
    }
    __syncthreads();   // the ONLY block barrier; K-loop below is barrier-free

    v8f acc0 = {}, acc1 = {}, acc2 = {}, acc3 = {};
    float dpart = 0.0f;

    #pragma unroll 2
    for (int k0 = 0; k0 < CIN; k0 += 32) {
        // speculative prefetch of next K-slice of fm (global_prefetch_b8)
        __builtin_prefetch(&img[imgBase + (long)(k0 + 32) * (WFULL * WFULL)], 0, 0);

        // ---- B fragment: 16 fm values per lane (column N = pos) ----
        v16bf bmat;
        float fmv[16];
        #pragma unroll
        for (int j = 0; j < 16; ++j) {
            const int kk = ((j < 8) ? 0 : 16) + khalf + (j & 7);
            const float v = img[imgBase + (long)(k0 + kk) * (WFULL * WFULL)];
            fmv[j]  = v;
            bmat[j] = (__bf16)v;
        }

        // ---- depth head: fp32 dot with the exact same K elements ----
        float dsum = 0.0f;
        #pragma unroll
        for (int j = 0; j < 16; ++j) {
            const int kk = ((j < 8) ? 0 : 16) + khalf + (j & 7);
            dsum += (float)a_lds[ENC * APITCH + k0 + kk] * fmv[j];
        }
        dpart += dsum;

        // ---- A fragments for 4 M-tiles (rows 0..63) from LDS ----
        v16bf a0, a1, a2, a3;
        #pragma unroll
        for (int j = 0; j < 16; ++j) {
            const int kk = ((j < 8) ? 0 : 16) + khalf + (j & 7);
            a0[j] = a_lds[(m0     ) * APITCH + k0 + kk];
            a1[j] = a_lds[(m0 + 16) * APITCH + k0 + kk];
            a2[j] = a_lds[(m0 + 32) * APITCH + k0 + kk];
            a3[j] = a_lds[(m0 + 48) * APITCH + k0 + kk];
        }

        acc0 = __builtin_amdgcn_wmma_f32_16x16x32_bf16(false, a0, false, bmat, (short)0, acc0, false, false);
        acc1 = __builtin_amdgcn_wmma_f32_16x16x32_bf16(false, a1, false, bmat, (short)0, acc1, false, false);
        acc2 = __builtin_amdgcn_wmma_f32_16x16x32_bf16(false, a2, false, bmat, (short)0, acc2, false, false);
        acc3 = __builtin_amdgcn_wmma_f32_16x16x32_bf16(false, a3, false, bmat, (short)0, acc3, false, false);
    }

    // ---- epilogue: bias, x = sum(pfm^2), stores ----
    const int  mtop  = (lane >> 4) * 8;    // 0 or 8 (which M half this lane holds)
    const long pbase = ((long)b * ENC) * NPIX + pos;
    float xs = 0.0f;
    #pragma unroll
    for (int r = 0; r < 8; ++r) {
        const int M0 = r + mtop;
        const float v0 = acc0[r] + b1[M0];
        const float v1 = acc1[r] + b1[M0 + 16];
        const float v2 = acc2[r] + b1[M0 + 32];
        const float v3 = acc3[r] + b1[M0 + 48];
        xs += v0 * v0 + v1 * v1 + v2 * v2 + v3 * v3;
        pfm[pbase + (long)(M0     ) * NPIX] = v0;
        pfm[pbase + (long)(M0 + 16) * NPIX] = v1;
        pfm[pbase + (long)(M0 + 32) * NPIX] = v2;
        pfm[pbase + (long)(M0 + 48) * NPIX] = v3;
    }
    // lanes L and L+16 hold complementary M halves of the same column N
    const float xfull = xs    + __shfl_xor(xs,    16, 32);
    const float dfull = dpart + __shfl_xor(dpart, 16, 32);
    if (lane < 16) {
        xmap[b * NPIX + pos]     = xfull;
        depthMap[b * NPIX + pos] = floorf(tanhf(dfull + bdep[0]) * ((float)HC / 10.0f));
    }
}

// =====================================================================
// K2: 5x5 soft-kernel correlations on x -> ordMean/absMean maps
// =====================================================================
__global__ __launch_bounds__(256)
void k_meanmaps(const float* __restrict__ xmap,
                float* __restrict__ ordMean, float* __restrict__ absMean)
{
    const int t = blockIdx.x * blockDim.x + threadIdx.x;
    if (t >= BATCH * NPIX) return;
    const int b = t / NPIX, p = t % NPIX;
    const int y = p / WC, x = p % WC;
    const float* xb = xmap + b * NPIX;

    float so = 0.0f, sa = 0.0f, sw = 0.0f;
    #pragma unroll
    for (int dy = -2; dy <= 2; ++dy) {
        #pragma unroll
        for (int dx = -2; dx <= 2; ++dx) {
            const int yy = y + dy, xx = x + dx;
            if (yy < 0 || yy >= HC || xx < 0 || xx >= WC) continue;
            const float w = 5.0f - (fabsf((float)dy) + fabsf((float)dx));
            const float v = xb[yy * WC + xx];
            so += (float)dy * w * v;
            sa += (float)dx * w * v;
            sw += w * v;
        }
    }
    const float inv = 1.0f / (sw + 0.001f);
    ordMean[t] = (float)y + so * inv;
    absMean[t] = (float)x + sa * inv;
}

// =====================================================================
// K3: per-batch top-1024 of 3136 via full bitonic sort (desc) in LDS
// =====================================================================
__global__ __launch_bounds__(1024)
void k_topk(const float* __restrict__ xmap, int* __restrict__ topidx)
{
    __shared__ float vals[NSORT];
    __shared__ int   idxs[NSORT];
    const int b = blockIdx.x;
    const int tid = threadIdx.x;

    for (int i = tid; i < NSORT; i += 1024) {
        vals[i] = (i < NPIX) ? xmap[b * NPIX + i] : -3.0e38f;
        idxs[i] = i;
    }
    __syncthreads();

    for (int k = 2; k <= NSORT; k <<= 1) {
        for (int j = k >> 1; j > 0; j >>= 1) {
            for (int i = tid; i < NSORT; i += 1024) {
                const int ixj = i ^ j;
                if (ixj > i) {
                    const bool descend = ((i & k) == 0);
                    const float v0 = vals[i], v1 = vals[ixj];
                    const int   i0 = idxs[i], i1 = idxs[ixj];
                    // "greater" with index tie-break (smaller index wins)
                    const bool gt = (v0 > v1) || (v0 == v1 && i0 < i1);
                    if (descend ? !gt : gt) {
                        vals[i] = v1; vals[ixj] = v0;
                        idxs[i] = i1; idxs[ixj] = i0;
                    }
                }
            }
            __syncthreads();
        }
    }
    for (int i = tid; i < KTOP; i += 1024) topidx[b * KTOP + i] = idxs[i];
}

// =====================================================================
// K4: gather + emit (points, batch, pos, pointfeatures) concatenated
// =====================================================================
__global__ __launch_bounds__(64)
void k_gather(const int* __restrict__ topidx,
              const float* __restrict__ ordMean, const float* __restrict__ absMean,
              const float* __restrict__ depthMap, const float* __restrict__ pfm,
              float* __restrict__ out)
{
    const int pair = blockIdx.x;              // 0 .. BATCH*KTOP-1
    const int b  = pair >> 10;
    const int ch = threadIdx.x;               // 0..63

    __shared__ float s_ab, s_od, s_dep;
    __shared__ int   s_idx2;
    if (ch == 0) {
        const int fidx = topidx[pair];
        const float od = ordMean[b * NPIX + fidx];
        const float ab = absMean[b * NPIX + fidx];
        int oi = (int)od; oi = oi < 0 ? 0 : (oi > HC - 1 ? HC - 1 : oi);   // trunc + clip
        int ai = (int)ab; ai = ai < 0 ? 0 : (ai > WC - 1 ? WC - 1 : ai);
        const int idx2 = oi * WC + ai;
        s_ab = ab; s_od = od; s_idx2 = idx2;
        s_dep = depthMap[b * NPIX + idx2];
    }
    __syncthreads();

    const float feat = pfm[(((long)b * ENC) + ch) * NPIX + s_idx2];

    float* points = out;                                        // (B,K,67)
    float* batchv = points + (long)BATCH * KTOP * (3 + ENC);    // (B*K)
    float* posv   = batchv + (long)BATCH * KTOP;                // (B*K,3)
    float* pfeat  = posv   + (long)BATCH * KTOP * 3;            // (B*K,ENC)

    const long pb = (long)pair * (3 + ENC);
    if (ch == 0) {
        points[pb + 0] = s_ab;
        points[pb + 1] = s_od;
        points[pb + 2] = s_dep;
        batchv[pair] = (float)b;
        posv[(long)pair * 3 + 0] = s_ab;
        posv[(long)pair * 3 + 1] = s_od;
        posv[(long)pair * 3 + 2] = s_dep;
    }
    points[pb + 3 + ch] = feat;
    pfeat[(long)pair * ENC + ch] = feat;
}

// =====================================================================
extern "C" void kernel_launch(void* const* d_in, const int* in_sizes, int n_in,
                              void* d_out, int out_size, void* d_ws, size_t ws_size,
                              hipStream_t stream) {
    (void)in_sizes; (void)n_in; (void)out_size; (void)ws_size;

    const float* img  = (const float*)d_in[0];   // imgBatch (16,1024,62,62)
    const float* w1   = (const float*)d_in[1];   // (64,1024)
    const float* b1   = (const float*)d_in[2];   // (64,)
    const float* wdep = (const float*)d_in[3];   // (1,1024)
    const float* bdep = (const float*)d_in[4];   // (1,)
    float* out = (float*)d_out;

    // workspace layout (floats)
    float* ws       = (float*)d_ws;
    float* pfm      = ws;                                   // B*ENC*NPIX
    float* depthMap = pfm + (size_t)BATCH * ENC * NPIX;     // B*NPIX
    float* xmap     = depthMap + (size_t)BATCH * NPIX;      // B*NPIX
    float* ordMean  = xmap     + (size_t)BATCH * NPIX;      // B*NPIX
    float* absMean  = ordMean  + (size_t)BATCH * NPIX;      // B*NPIX
    int*   topidx   = (int*)(absMean + (size_t)BATCH * NPIX); // B*KTOP ints

    // K1: fused WMMA GEMM (3136 tiles, 8 waves/block)
    k_gemm_fused<<<(BATCH * NTILE) / 8, 256, 0, stream>>>(
        img, w1, b1, wdep, bdep, pfm, depthMap, xmap);

    // K2: ordMean/absMean
    k_meanmaps<<<(BATCH * NPIX + 255) / 256, 256, 0, stream>>>(
        xmap, ordMean, absMean);

    // K3: top-k per batch
    k_topk<<<BATCH, 1024, 0, stream>>>(xmap, topidx);

    // K4: gather + outputs
    k_gather<<<BATCH * KTOP, 64, 0, stream>>>(
        topidx, ordMean, absMean, depthMap, pfm, out);
}
